// DeepSeekV3Router_19413252178049
// MI455X (gfx1250) — matrix operand order repaired and checked by reading, not compiled, not run on hardware
//
#include <hip/hip_runtime.h>
#include <hip/hip_bf16.h>

typedef __attribute__((ext_vector_type(16))) __bf16 v16bf;
typedef __attribute__((ext_vector_type(8)))  float  v8f;

#define DIM   7168
#define NEXP  256
#define NKC   (DIM / 32)     // 224 K-chunks of 32

// ---------- bf16 helpers ----------
__device__ __forceinline__ unsigned short f2bf_rne(float f) {   // one-time prep only
    unsigned u = __float_as_uint(f);
    unsigned r = u + 0x7FFFu + ((u >> 16) & 1u);
    return (unsigned short)(r >> 16);
}
__device__ __forceinline__ float bf2f(unsigned short h) {
    return __uint_as_float(((unsigned)h) << 16);
}

// Cheap truncation split for the hot loop: hi = top16(f) (exact bf16 by truncation),
// lo = f - hi is exact in f32 (Sterbenz), then truncated to bf16 (err ~2^-16 rel).
// Packs two elements per dword with a single v_perm_b32 each for hi and lo.
__device__ __forceinline__ void split2(float f0, float f1, unsigned& hipack, unsigned& lopack) {
    unsigned u0 = __float_as_uint(f0);
    unsigned u1 = __float_as_uint(f1);
    hipack = __builtin_amdgcn_perm(u1, u0, 0x07060302u);   // {u1.hi16, u0.hi16}
    float l0 = f0 - __uint_as_float(u0 & 0xffff0000u);
    float l1 = f1 - __uint_as_float(u1 & 0xffff0000u);
    lopack = __builtin_amdgcn_perm(__float_as_uint(l1), __float_as_uint(l0), 0x07060302u);
}

union V16BF {
    unsigned short s[16];
    unsigned       d[8];
    uint4          u4[2];
    v16bf          v;
};

// ---------- Kernel 1: repack kernel_DE (f32, DxE row-major) into WMMA B-layout bf16 hi/lo ----------
// B 32x16 (KxN), this code's k order: lane L holds column N = L%16; element e (0..15) is K = (L/16)*16 + e.
// Flat: idx = (((kc*16 + tn)*32) + lane)*16 + e -> per-lane 32B contiguous, lanes contiguous.
__global__ __launch_bounds__(256) void prep_b_kernel(const float* __restrict__ kDE,
                                                     unsigned short* __restrict__ bhi,
                                                     unsigned short* __restrict__ blo) {
    int idx = blockIdx.x * 256 + threadIdx.x;
    if (idx >= DIM * NEXP) return;
    int e    = idx & 15;
    int lane = (idx >> 4) & 31;
    int tn   = (idx >> 9) & 15;
    int kc   = idx >> 13;
    int k = kc * 32 + (lane >> 4) * 16 + e;
    int n = tn * 16 + (lane & 15);
    float v = kDE[(size_t)k * NEXP + n];
    unsigned short h = f2bf_rne(v);
    bhi[idx] = h;
    blo[idx] = f2bf_rne(v - bf2f(h));
}

// ---------- Kernel 2: gate GEMM (bf16 hi/lo split, 3 WMMA per tile) + fused sigmoid ----------
// Block = 8 waves covering 32 rows x 256 experts.
// Wave w: rows 16*(w>>2) .. +16, col tiles 4*(w&3) .. +4  -> 4 accumulators, 12 WMMA / K-chunk.
__global__ __launch_bounds__(256) void gate_gemm_kernel(const float* __restrict__ x,
                                                        const uint4* __restrict__ bhi4,
                                                        const uint4* __restrict__ blo4,
                                                        float* __restrict__ scores) {
    const int wave    = threadIdx.x >> 5;
    const int lane    = threadIdx.x & 31;
    const int hl      = lane >> 4;
    const int ln16    = lane & 15;
    const int tm      = blockIdx.x;
    const int rowHalf = wave >> 2;
    const int tnBase  = (wave & 3) * 4;

    // A 16x32 bf16, this code's k order (same as B): lane%16 = M; lanes0-15 els0..15=K0..15;
    // lanes16-31 els0..15=K16..31.
    const float* xrow = x + (size_t)(tm * 32 + rowHalf * 16 + ln16) * DIM;

    v8f acc[4];
    #pragma unroll
    for (int t = 0; t < 4; ++t) acc[t] = (v8f){0.f,0.f,0.f,0.f,0.f,0.f,0.f,0.f};

    for (int kc = 0; kc < NKC; ++kc) {
        const int kb = kc * 32;
        float4 a0 = *(const float4*)(xrow + kb + hl * 16);
        float4 a1 = *(const float4*)(xrow + kb + hl * 16 + 4);
        float4 a2 = *(const float4*)(xrow + kb + hl * 16 + 8);
        float4 a3 = *(const float4*)(xrow + kb + hl * 16 + 12);
        if (kc + 1 < NKC) __builtin_prefetch(xrow + kb + 32, 0, 0);  // global_prefetch_b8

        float af[16] = {a0.x,a0.y,a0.z,a0.w, a1.x,a1.y,a1.z,a1.w,
                        a2.x,a2.y,a2.z,a2.w, a3.x,a3.y,a3.z,a3.w};
        V16BF ahi, alo;
        #pragma unroll
        for (int j = 0; j < 8; ++j)
            split2(af[2*j], af[2*j+1], ahi.d[j], alo.d[j]);

        V16BF bh[4], bl[4];
        #pragma unroll
        for (int t = 0; t < 4; ++t) {
            const size_t bofs = ((size_t)(kc * 16 + tnBase + t) * 32 + lane) * 2;  // uint4 units
            bh[t].u4[0] = bhi4[bofs];  bh[t].u4[1] = bhi4[bofs + 1];
            bl[t].u4[0] = blo4[bofs];  bl[t].u4[1] = blo4[bofs + 1];
        }

        // D += Ahi*Bhi + Ahi*Blo + Alo*Bhi; same-acc WMMA distance = 3 for pipelining
        #pragma unroll
        for (int t = 0; t < 4; ++t)
            acc[t] = __builtin_amdgcn_wmma_f32_16x16x32_bf16(false, ahi.v, false, bh[t].v, (short)0, acc[t], false, false);
        #pragma unroll
        for (int t = 0; t < 4; ++t)
            acc[t] = __builtin_amdgcn_wmma_f32_16x16x32_bf16(false, ahi.v, false, bl[t].v, (short)0, acc[t], false, false);
        #pragma unroll
        for (int t = 0; t < 4; ++t)
            acc[t] = __builtin_amdgcn_wmma_f32_16x16x32_bf16(false, alo.v, false, bh[t].v, (short)0, acc[t], false, false);
    }

    // C/D layout: M = r + hl*8, N = lane%16. Fuse sigmoid.
    #pragma unroll
    for (int t = 0; t < 4; ++t) {
        #pragma unroll
        for (int r = 0; r < 8; ++r) {
            int row = tm * 32 + rowHalf * 16 + hl * 8 + r;
            int col = (tnBase + t) * 16 + ln16;
            scores[(size_t)row * NEXP + col] = 1.0f / (1.0f + __expf(-acc[t][r]));
        }
    }
}

// ---------- Kernel 3: grouped top-k routing, one wave32 per token ----------
__global__ __launch_bounds__(256) void router_kernel(const float* __restrict__ scores,
                                                     const float* __restrict__ bias,
                                                     float* __restrict__ out,
                                                     int T) {
    const int lane = threadIdx.x & 31;
    const int t    = blockIdx.x * 8 + (threadIdx.x >> 5);
    if (t >= T) return;

    // Each lane owns experts [lane*8, lane*8+8); group g = experts 32g..32g+31 = lanes 4g..4g+3.
    const float4* sp = (const float4*)(scores + (size_t)t * NEXP + lane * 8);
    const float4* bp = (const float4*)(bias + lane * 8);
    float4 s0 = sp[0], s1 = sp[1];
    float4 b0 = bp[0], b1 = bp[1];

    float s[8]  = {s0.x, s0.y, s0.z, s0.w, s1.x, s1.y, s1.z, s1.w};              // original sigmoid scores
    float bb[8] = {s0.x + b0.x, s0.y + b0.y, s0.z + b0.z, s0.w + b0.w,
                   s1.x + b1.x, s1.y + b1.y, s1.z + b1.z, s1.w + b1.w};          // biased scores

    // per-lane top-2 of biased
    float m1 = -1e30f, m2 = -1e30f;
    #pragma unroll
    for (int e = 0; e < 8; ++e) {
        float v = bb[e];
        if (v > m1) { m2 = m1; m1 = v; }
        else if (v > m2) { m2 = v; }
    }
    // merge top-2 across the 4 lanes of each group (xor 1, xor 2 stay in-group)
    #pragma unroll
    for (int off = 1; off <= 2; off <<= 1) {
        float o1 = __shfl_xor(m1, off, 32);
        float o2 = __shfl_xor(m2, off, 32);
        float n1 = fmaxf(m1, o1);
        float n2 = fmaxf(fminf(m1, o1), fmaxf(m2, o2));
        m1 = n1; m2 = n2;
    }
    float gs  = m1 + m2;      // my group's score
    int   myg = lane >> 2;

    // top-4 groups by rank (ties -> lower group index wins, matching lax.top_k)
    int rank = 0;
    #pragma unroll
    for (int g = 0; g < 8; ++g) {
        float og = __shfl(gs, g * 4, 32);
        rank += (og > gs || (og == gs && g < myg)) ? 1 : 0;
    }
    const bool selected = (rank < 4);

    float vals[8];
    #pragma unroll
    for (int e = 0; e < 8; ++e) vals[e] = selected ? bb[e] : 0.0f;

    // iterative top-8 with (value desc, index asc) ordering; carry original score for the gather
    float w[8]; int wi[8]; float wsum = 0.0f;
    #pragma unroll
    for (int it = 0; it < 8; ++it) {
        float bv = vals[0]; int be = 0; float bor = s[0];
        #pragma unroll
        for (int e = 1; e < 8; ++e)
            if (vals[e] > bv) { bv = vals[e]; be = e; bor = s[e]; }
        int bidx = lane * 8 + be;
        #pragma unroll
        for (int off = 16; off >= 1; off >>= 1) {
            float ov = __shfl_xor(bv, off, 32);
            int   oi = __shfl_xor(bidx, off, 32);
            float oo = __shfl_xor(bor, off, 32);
            if (ov > bv || (ov == bv && oi < bidx)) { bv = ov; bidx = oi; bor = oo; }
        }
        w[it] = bor; wi[it] = bidx; wsum += bor;
        // knock out the winner (constant-index writes only)
        bool mine = ((bidx >> 3) == lane);
        int  rem  = bidx & 7;
        #pragma unroll
        for (int e = 0; e < 8; ++e)
            vals[e] = (mine && e == rem) ? -1e30f : vals[e];
    }

    if (lane == 0) {
        float scale = 2.5f / (wsum + 1e-20f);
        #pragma unroll
        for (int i = 0; i < 8; ++i) {
            out[(size_t)t * 8 + i]                 = w[i] * scale;   // weights block
            out[(size_t)T * 8 + (size_t)t * 8 + i] = (float)wi[i];   // indices block (flat-concat)
        }
    }
}

// ---------------------------------------------------------------------------
extern "C" void kernel_launch(void* const* d_in, const int* in_sizes, int n_in,
                              void* d_out, int out_size, void* d_ws, size_t ws_size,
                              hipStream_t stream) {
    const float* x    = (const float*)d_in[0];  // [T, 7168] f32
    const float* kDE  = (const float*)d_in[1];  // [7168, 256] f32
    const float* bias = (const float*)d_in[2];  // [256] f32
    float* out = (float*)d_out;

    const int T = in_sizes[0] / DIM;            // 16384

    // workspace layout: Bhi | Blo | scores
    char* ws = (char*)d_ws;
    unsigned short* bhi = (unsigned short*)ws;                       // DIM*NEXP bf16 = 3.67 MB
    unsigned short* blo = bhi + (size_t)DIM * NEXP;                  // 3.67 MB
    float* scoresBuf    = (float*)(ws + 2 * (size_t)DIM * NEXP * 2); // T*256 f32 = 16.8 MB

    prep_b_kernel<<<(DIM * NEXP + 255) / 256, 256, 0, stream>>>(kDE, bhi, blo);
    gate_gemm_kernel<<<T / 32, 256, 0, stream>>>(x, (const uint4*)bhi, (const uint4*)blo, scoresBuf);
    router_kernel<<<(T + 7) / 8, 256, 0, stream>>>(scoresBuf, bias, out, T);
}
